// GraphConv_5488968204378
// MI455X (gfx1250) — matrix operand order repaired and checked by reading, not compiled, not run on hardware
//
#include <hip/hip_runtime.h>
#include <hip/hip_bf16.h>
#include <stdint.h>

// ---------------------------------------------------------------------------
// GCN layer: out = (D^-1/2 (A+I) D^-1/2) X W^T + b
// Refactored: h[i,:] = d[i]*(A[i,:] @ (d .* x)) + d[i]^2 * x[i,:]
//             out    = h @ W^T + b
// Big GEMM runs in BF16 WMMA (v_wmma_f32_16x16x32_bf16) with f32 accumulate,
// software-pipelined global->reg->LDS staging.
// ---------------------------------------------------------------------------

typedef __attribute__((ext_vector_type(16))) __bf16 v16bf;
typedef __attribute__((ext_vector_type(8)))  float  v8f;

// Branchless float->bf16 round-to-nearest-even (inputs are finite here).
__device__ __forceinline__ unsigned short f2bf_rne(float f) {
  const unsigned int u = __float_as_uint(f);
  return (unsigned short)((u + 0x7fffu + ((u >> 16) & 1u)) >> 16);
}

// ---------------- kernel 1: d[i] = rsqrt(rowsum(adj) + 1) ------------------
__global__ __launch_bounds__(256) void gcn_rowsum_kernel(
    const float* __restrict__ adj, float* __restrict__ dvec, int N) {
  const int row = blockIdx.x;
  const float* p = adj + (size_t)row * N;
  float s = 0.f;
  for (int c = threadIdx.x * 4; c < N; c += 256 * 4) {
    const float4 v = *(const float4*)(p + c);
    s += v.x + v.y + v.z + v.w;
  }
  __shared__ float red[256];
  red[threadIdx.x] = s;
  __syncthreads();
  for (int off = 128; off > 0; off >>= 1) {
    if (threadIdx.x < off) red[threadIdx.x] += red[threadIdx.x + off];
    __syncthreads();
  }
  if (threadIdx.x == 0) dvec[row] = rsqrtf(red[0] + 1.0f);
}

// -------- kernel 2: xdT[f][j] = bf16(d[j] * x[j][f])  (32x32 LDS tiles) ----
__global__ __launch_bounds__(256) void gcn_scale_transpose_kernel(
    const float* __restrict__ x, const float* __restrict__ dvec,
    unsigned short* __restrict__ xdT, int N, int F) {
  __shared__ float t[32][33];
  const int tx = threadIdx.x & 31;
  const int ty = threadIdx.x >> 5;                 // 0..7
  const int j0 = blockIdx.y * 32;
  const int f0 = blockIdx.x * 32;
#pragma unroll
  for (int i = 0; i < 4; ++i) {
    const int jl = ty + i * 8;
    t[jl][tx] = x[(size_t)(j0 + jl) * F + f0 + tx];   // coalesced read
  }
  __syncthreads();
  const float dj = dvec[j0 + tx];
#pragma unroll
  for (int i = 0; i < 4; ++i) {
    const int fl = ty + i * 8;
    xdT[(size_t)(f0 + fl) * N + j0 + tx] = f2bf_rne(dj * t[tx][fl]); // coalesced write
  }
}

// ---------------- kernel 3: W (fp32) -> W (bf16), same layout --------------
__global__ __launch_bounds__(256) void gcn_wconv_kernel(
    const float* __restrict__ W, unsigned short* __restrict__ Wb) {
  const int idx = (blockIdx.x * 256 + threadIdx.x) * 4;
  const float4 v = *(const float4*)(W + idx);
  union { unsigned short h[4]; uint2 u; } p;
  p.h[0] = f2bf_rne(v.x); p.h[1] = f2bf_rne(v.y);
  p.h[2] = f2bf_rne(v.z); p.h[3] = f2bf_rne(v.w);
  *(uint2*)(Wb + idx) = p.u;
}

// -------------------------- WMMA BF16 GEMM ---------------------------------
// C[M x Ntot] = A[M x K] * B^T  where B is stored row-major [Ntot][K].
// Block: 128 threads = 4 waves (2x2), block tile 128(M) x 64(N), K-step 64.
// Wave tile 64(M) x 32(N) = 4x2 WMMA tiles -> 8 v_wmma per 32-wide k-chunk.
// Software pipeline: global loads for tile k+1 issue before the WMMA burst
// on tile k; the s_wait_loadcnt lands after compute+barrier.
// MODE 0: A = adj (fp32 -> bf16 while staging); epilogue
//         h_bf16[i][f] = bf16(d[i]*acc + d[i]^2 * x[i][f])
// MODE 1: A = h (bf16); epilogue out[i][o] = acc + b[o]  (fp32)
#define TM   128
#define TN   64
#define TK   64
#define LDSK 80   // +16 bf16 pad: 160B row stride, 16B aligned, conflict-broken

template <int MODE>
__global__ __launch_bounds__(128) void gcn_gemm_kernel(
    const void* __restrict__ Aptr,
    const unsigned short* __restrict__ Bptr,   // [Ntot][K] bf16 row-major
    const float* __restrict__ dvec,
    const float* __restrict__ xres,
    const float* __restrict__ bias,
    unsigned short* __restrict__ outBf,
    float* __restrict__ outF,
    int M, int Ntot, int K) {
  __shared__ __align__(16) unsigned short As[TM * LDSK];  // 20 KB
  __shared__ __align__(16) unsigned short Bs[TN * LDSK];  // 10 KB

  const int tid   = threadIdx.x;
  const int lane  = tid & 31;
  const int wave  = tid >> 5;     // 4 waves
  const int waveM = wave >> 1;    // 0..1 -> M offset 0/64
  const int waveN = wave & 1;     // 0..1 -> N offset 0/32
  const int lr    = lane & 15;
  const int hsel  = lane >> 4;    // half-wave select

  const int m0 = blockIdx.y * TM;
  const int n0 = blockIdx.x * TN;

  v8f zero = {};
  v8f acc[4][2];
#pragma unroll
  for (int m = 0; m < 4; ++m) { acc[m][0] = zero; acc[m][1] = zero; }

  // register staging buffers (one K-step ahead)
  float4 aRegF[16];   // MODE 0: 128x64 fp32 tile / 128 threads
  uint4  aRegH[8];    // MODE 1: 128x64 bf16 tile / 128 threads
  uint4  bReg[4];     //         64x64  bf16 tile / 128 threads

  auto load_tile = [&](int k0) {
    if constexpr (MODE == 0) {
      const float* A = (const float*)Aptr;
      const int cg = tid & 15;     // float4 column group (16 per row)
      const int r0 = tid >> 4;     // 0..7
#pragma unroll
      for (int rep = 0; rep < 16; ++rep) {
        const int row = r0 + rep * 8;
        aRegF[rep] = *(const float4*)(A + (size_t)(m0 + row) * K + k0 + cg * 4);
      }
    } else {
      const unsigned short* A = (const unsigned short*)Aptr;
      const int g  = tid & 7;      // uint4 group (8 per row)
      const int r0 = tid >> 3;     // 0..15
#pragma unroll
      for (int rep = 0; rep < 8; ++rep) {
        const int row = r0 + rep * 16;
        aRegH[rep] = *(const uint4*)(A + (size_t)(m0 + row) * K + k0 + g * 8);
      }
    }
    {
      const int g  = tid & 7;
      const int r0 = tid >> 3;     // 0..15
#pragma unroll
      for (int rep = 0; rep < 4; ++rep) {
        const int row = r0 + rep * 16;
        bReg[rep] = *(const uint4*)(Bptr + (size_t)(n0 + row) * K + k0 + g * 8);
      }
    }
  };

  auto store_tile = [&]() {
    if constexpr (MODE == 0) {
      const int cg = tid & 15;
      const int r0 = tid >> 4;
#pragma unroll
      for (int rep = 0; rep < 16; ++rep) {
        const int row = r0 + rep * 8;
        union { unsigned short h[4]; uint2 u; } p;
        p.h[0] = f2bf_rne(aRegF[rep].x); p.h[1] = f2bf_rne(aRegF[rep].y);
        p.h[2] = f2bf_rne(aRegF[rep].z); p.h[3] = f2bf_rne(aRegF[rep].w);
        *(uint2*)&As[row * LDSK + cg * 4] = p.u;
      }
    } else {
      const int g  = tid & 7;
      const int r0 = tid >> 3;
#pragma unroll
      for (int rep = 0; rep < 8; ++rep) {
        const int row = r0 + rep * 16;
        *(uint4*)&As[row * LDSK + g * 8] = aRegH[rep];
      }
    }
    {
      const int g  = tid & 7;
      const int r0 = tid >> 3;
#pragma unroll
      for (int rep = 0; rep < 4; ++rep) {
        const int row = r0 + rep * 16;
        *(uint4*)&Bs[row * LDSK + g * 8] = bReg[rep];
      }
    }
  };

  load_tile(0);

  for (int k0 = 0; k0 < K; k0 += TK) {
    store_tile();          // waits on loads issued one step ago
    __syncthreads();

    if (k0 + TK < K) {
      load_tile(k0 + TK);  // issue next tile; no wait until next store_tile
      if (k0 + 2 * TK < K) {
        const char* pf = (MODE == 0)
            ? (const char*)((const float*)Aptr + (size_t)(m0 + tid) * K + (k0 + 2 * TK))
            : (const char*)((const unsigned short*)Aptr + (size_t)(m0 + tid) * K + (k0 + 2 * TK));
        __builtin_prefetch(pf, 0, 1);  // global_prefetch_b8, 2 tiles ahead
      }
    }

#pragma unroll
    for (int kc = 0; kc < TK; kc += 32) {
      union { v16bf v; uint4 q[2]; } fa[4], fb[2];
      // A frag (16x32 bf16): lane<16 holds K {0..7,16..23}; lane>=16 {8..15,24..31}
#pragma unroll
      for (int m = 0; m < 4; ++m) {
        const int row = waveM * 64 + m * 16 + lr;
        const unsigned short* base = &As[row * LDSK + kc + hsel * 8];
        fa[m].q[0] = *(const uint4*)(base);
        fa[m].q[1] = *(const uint4*)(base + 16);
      }
      // B frag (32x16 bf16): lane<16 holds K {0..7,16..23}; lane>=16 {8..15,24..31}
#pragma unroll
      for (int n = 0; n < 2; ++n) {
        const int row = waveN * 32 + n * 16 + lr;
        const unsigned short* base = &Bs[row * LDSK + kc + hsel * 8];
        fb[n].q[0] = *(const uint4*)(base);
        fb[n].q[1] = *(const uint4*)(base + 16);
      }
#pragma unroll
      for (int m = 0; m < 4; ++m)
#pragma unroll
        for (int n = 0; n < 2; ++n)
          acc[m][n] = __builtin_amdgcn_wmma_f32_16x16x32_bf16(
              false, fa[m].v, false, fb[n].v, (short)0, acc[m][n], false, false);
    }
    __syncthreads();
  }

  // ---- epilogue (C layout: VGPR r -> M = r + 8*hsel, N = lane&15) ----
#pragma unroll
  for (int m = 0; m < 4; ++m) {
    const int rbase = m0 + waveM * 64 + m * 16 + hsel * 8;
#pragma unroll
    for (int n = 0; n < 2; ++n) {
      const int col = n0 + waveN * 32 + n * 16 + lr;
      v8f a = acc[m][n];
      if (MODE == 0) {
#pragma unroll
        for (int r = 0; r < 8; ++r) {
          const int i = rbase + r;
          const float di = dvec[i];
          const float v = di * a[r] + di * di * xres[(size_t)i * Ntot + col];
          outBf[(size_t)i * Ntot + col] = f2bf_rne(v);
        }
      } else {
        const float bo = bias[col];
#pragma unroll
        for (int r = 0; r < 8; ++r)
          outF[(size_t)(rbase + r) * Ntot + col] = a[r] + bo;
      }
    }
  }
}

// ---------------------------------------------------------------------------
extern "C" void kernel_launch(void* const* d_in, const int* in_sizes, int n_in,
                              void* d_out, int out_size, void* d_ws, size_t ws_size,
                              hipStream_t stream) {
  const int N = 4096, F = 512, O = 512;

  const float* x   = (const float*)d_in[0];  // [N, F]
  const float* adj = (const float*)d_in[1];  // [N, N]
  const float* W   = (const float*)d_in[2];  // [O, F]
  const float* b   = (const float*)d_in[3];  // [O]
  float* out = (float*)d_out;                // [N, O]

  char* ws = (char*)d_ws;
  float*          dvec = (float*)ws;                                       // 16 KB
  unsigned short* xdT  = (unsigned short*)(ws + (16u << 10));              // [F][N] bf16, 4 MB
  unsigned short* hb   = (unsigned short*)(ws + (16u << 10) + (4u << 20)); // [N][F] bf16, 4 MB
  unsigned short* Wb   = (unsigned short*)(ws + (16u << 10) + (8u << 20)); // [O][F] bf16, 512 KB

  // 1) degree vector
  gcn_rowsum_kernel<<<N, 256, 0, stream>>>(adj, dvec, N);

  // 2) xdT[f][j] = bf16(d[j] * x[j][f])   (B operand for GEMM1, row-major [F][N])
  dim3 gT(F / 32, N / 32);
  gcn_scale_transpose_kernel<<<gT, 256, 0, stream>>>(x, dvec, xdT, N, F);

  // 3) W -> bf16 (B operand for GEMM2, already [O][F] = row-major [N][K])
  gcn_wconv_kernel<<<(O * F) / 1024, 256, 0, stream>>>(W, Wb);

  // 4) h = d[i]*(adj @ xdT^T) + d[i]^2*x   -> bf16 [N][F]
  dim3 g1(F / TN, N / TM);   // (8, 32)
  gcn_gemm_kernel<0><<<g1, 128, 0, stream>>>(
      (const void*)adj, xdT, dvec, x, nullptr, hb, nullptr, N, F, N);

  // 5) out = h @ W^T + b  -> fp32 [N][O]
  dim3 g2(O / TN, N / TM);   // (8, 32)
  gcn_gemm_kernel<1><<<g2, 128, 0, stream>>>(
      (const void*)hb, Wb, nullptr, nullptr, b, nullptr, out, N, O, F);
}
